// LlamaAttention_heavy_hitter_8108898255541
// MI455X (gfx1250) — matrix operand order repaired and checked by reading, not compiled, not run on hardware
//
#include <hip/hip_runtime.h>
#include <math.h>

// ---------------- problem constants ----------------
#define Bz   2
#define Sz   1024
#define HIDz 2048
#define NHz  16
#define HDz  128
#define SBz  20      // floor(0.02*1024+0.5)
#define SELz 61      // floor(0.06*1024+0.5)
#define RBz  82      // floor(0.08*1024+0.5)
#define CBz  (SBz + SELz + RBz)   // 163
#define FFc  0.9f
#define MASKVAL (-3.402823466e38f)
#define SCALEc  0.08838834764831845f   // 1/sqrt(128)
#define KCz  128     // K-chunk staged in LDS by the TDM

typedef __attribute__((ext_vector_type(16))) __bf16 v16bf;
typedef __attribute__((ext_vector_type(8)))  float  v8f;
typedef __attribute__((ext_vector_type(4)))  unsigned int v4u;
typedef __attribute__((ext_vector_type(8)))  int v8i;
typedef __attribute__((ext_vector_type(4)))  int v4i;

union FragU { uint4 u[2]; v16bf v; };

__device__ __forceinline__ v16bf ld_frag16(const __bf16* p0, const __bf16* p1) {
  FragU f;
  f.u[0] = *(const uint4*)p0;
  f.u[1] = *(const uint4*)p1;
  return f.v;
}

// ---------------- wave32 reductions ----------------
__device__ __forceinline__ float wredsum(float v) {
#pragma unroll
  for (int o = 16; o > 0; o >>= 1) v += __shfl_xor(v, o, 32);
  return v;
}
__device__ __forceinline__ float wredmax(float v) {
#pragma unroll
  for (int o = 16; o > 0; o >>= 1) v = fmaxf(v, __shfl_xor(v, o, 32));
  return v;
}
__device__ __forceinline__ void wredargmin(float& v, int& i) {
#pragma unroll
  for (int o = 16; o > 0; o >>= 1) {
    float ov = __shfl_xor(v, o, 32);
    int   oi = __shfl_xor(i, o, 32);
    if (ov < v || (ov == v && oi < i)) { v = ov; i = oi; }
  }
}

// 1024-thread block reductions: 32 waves -> exactly 32 partials -> every wave
// re-reduces all partials (no broadcast needed).
__device__ __forceinline__ float block_sum_1024(float v, float* sh) {
  int lane = threadIdx.x & 31, wid = threadIdx.x >> 5;
  v = wredsum(v);
  if (lane == 0) sh[wid] = v;
  __syncthreads();
  float r = sh[lane];
  r = wredsum(r);
  __syncthreads();
  return r;
}
__device__ __forceinline__ void block_argmin_1024(float& v, int& i, float* shv, int* shi) {
  int lane = threadIdx.x & 31, wid = threadIdx.x >> 5;
  wredargmin(v, i);
  if (lane == 0) { shv[wid] = v; shi[wid] = i; }
  __syncthreads();
  float rv = shv[lane]; int ri = shi[lane];
  wredargmin(rv, ri);
  __syncthreads();
  v = rv; i = ri;
}

// 256-thread (8-wave) block reductions for softmax rows
__device__ __forceinline__ float block_max_256(float v, float* sh) {
  int lane = threadIdx.x & 31, wid = threadIdx.x >> 5;
  v = wredmax(v);
  if (lane == 0) sh[wid] = v;
  __syncthreads();
  float r = (lane < 8) ? sh[lane] : -__builtin_inff();
  r = wredmax(r);
  __syncthreads();
  return r;
}
__device__ __forceinline__ float block_sum_256(float v, float* sh) {
  int lane = threadIdx.x & 31, wid = threadIdx.x >> 5;
  v = wredsum(v);
  if (lane == 0) sh[wid] = v;
  __syncthreads();
  float r = (lane < 8) ? sh[lane] : 0.0f;
  r = wredsum(r);
  __syncthreads();
  return r;
}

// ---------------- WMMA 16x64 strip core (direct-global B) ----------------
// A: [16 x K] row-major bf16 (pre-offset to strip row), lda = K
// Bt: [64 x K] row-major bf16 (B transposed; pre-offset to strip col), ldb = K
// All four B fragments are loaded into distinct registers BEFORE the WMMA
// chain so load waits can be staggered instead of serializing each WMMA.
__device__ __forceinline__ void wmma_strip(const __bf16* __restrict__ A,
                                           const __bf16* __restrict__ Bt,
                                           int K, v8f acc[4]) {
  const int lane = threadIdx.x & 31;
  const int mr   = lane & 15;
  const int half = lane >> 4;
  const __bf16* ap  = A  + (size_t)mr * K + half * 16;
  const __bf16* bp0 = Bt + (size_t)mr * K + half * 16;
  for (int k0 = 0; k0 < K; k0 += 32) {
    v16bf fa = ld_frag16(ap + k0, ap + k0 + 8);
    const __bf16* b0 = bp0 + k0;
    const __bf16* b1 = bp0 + (size_t)16 * K + k0;
    const __bf16* b2 = bp0 + (size_t)32 * K + k0;
    const __bf16* b3 = bp0 + (size_t)48 * K + k0;
    v16bf fb0 = ld_frag16(b0, b0 + 8);
    v16bf fb1 = ld_frag16(b1, b1 + 8);
    v16bf fb2 = ld_frag16(b2, b2 + 8);
    v16bf fb3 = ld_frag16(b3, b3 + 8);
    acc[0] = __builtin_amdgcn_wmma_f32_16x16x32_bf16(false, fa, false, fb0, (short)0, acc[0], false, false);
    acc[1] = __builtin_amdgcn_wmma_f32_16x16x32_bf16(false, fa, false, fb1, (short)0, acc[1], false, false);
    acc[2] = __builtin_amdgcn_wmma_f32_16x16x32_bf16(false, fa, false, fb2, (short)0, acc[2], false, false);
    acc[3] = __builtin_amdgcn_wmma_f32_16x16x32_bf16(false, fa, false, fb3, (short)0, acc[3], false, false);
  }
}

// ---------------- TDM: stage a 64 x KC bf16 tile of Bt into LDS ----------------
// Builds a 2D Tensor DMA Descriptor (ISA cdna5 §8) and issues TENSOR_LOAD_TO_LDS.
__device__ __forceinline__ void tdm_issue_b_tile(const __bf16* gptr, int ldK,
                                                 void* lptr) {
  unsigned lds = (unsigned)(size_t)lptr;                  // low 32b = LDS byte addr
  unsigned long long ga = (unsigned long long)(size_t)gptr;
  unsigned td0 = (unsigned)ldK;                           // tensor_dim0 (elements)
  unsigned td1 = 64u;                                     // tensor_dim1 (rows)
  unsigned long long st0 = (unsigned long long)ldK;       // dim0 stride (elements)
  v4u g0;
  g0[0] = 1u;                                             // count=1, user mode
  g0[1] = lds;                                            // lds_addr
  g0[2] = (unsigned)(ga & 0xFFFFFFFFu);                   // global_addr[31:0]
  g0[3] = (unsigned)((ga >> 32) & 0x01FFFFFFu) | (2u << 30); // addr[56:32] | type=2
  v8i g1;
  g1[0] = (int)(1u << 16);                                // data_size=1 (2 bytes)
  g1[1] = (int)((td0 & 0xFFFFu) << 16);                   // tensor_dim0 lo16
  g1[2] = (int)(((td0 >> 16) & 0xFFFFu) | ((td1 & 0xFFFFu) << 16)); // td0 hi | td1 lo
  g1[3] = (int)(((td1 >> 16) & 0xFFFFu) | ((unsigned)KCz << 16));   // td1 hi | tile_dim0
  g1[4] = (int)64u;                                       // tile_dim1=64, tile_dim2=0
  g1[5] = (int)(unsigned)(st0 & 0xFFFFFFFFu);             // dim0_stride lo32
  g1[6] = (int)(unsigned)((st0 >> 32) & 0xFFFFu);         // dim0_stride hi16
  g1[7] = 0;                                              // dim1_stride (unused, 2D)
  v4i z4 = {0, 0, 0, 0};
  v8i z8 = {0, 0, 0, 0, 0, 0, 0, 0};
  __builtin_amdgcn_tensor_load_to_lds(g0, g1, z4, z4, z8, 0);
}

// ---------------- conversion kernels ----------------
__global__ void k_cvt_transpose(const float* __restrict__ w, __bf16* __restrict__ wt) {
  int tid = blockIdx.x * 256 + threadIdx.x;       // over 2048*2048
  int k = tid >> 11;
  int n = tid & 2047;
  wt[(size_t)n * HIDz + k] = (__bf16)w[tid];
}
__global__ void k_cvt(const float* __restrict__ x, __bf16* __restrict__ y) {
  int tid = blockIdx.x * 256 + threadIdx.x;
  y[tid] = (__bf16)x[tid];
}

// ---------------- generic GEMM with TDM-staged B tile ----------------
// C_f32[M,N] = A_bf16[M,K] @ Bt_bf16[N,K]^T.  grid(N/64, M/128), 256 threads.
// The 64-column B tile is DMA'd to LDS per K-chunk (double buffered) by the
// Tensor Data Mover; waves read B fragments from LDS, A straight from global.
__global__ void k_gemm(const __bf16* __restrict__ A, const __bf16* __restrict__ Bt,
                       float* __restrict__ C, int N, int K) {
  __shared__ __align__(16) __bf16 sB[2][64 * KCz];   // 2 x 16 KiB
  int wid = threadIdx.x >> 5;
  int lane = threadIdx.x & 31;
  int r0 = blockIdx.y * 128 + wid * 16;
  int c0 = blockIdx.x * 64;
  const __bf16* btile = Bt + (size_t)c0 * K;
  const int mr = lane & 15, half = lane >> 4;
  const __bf16* ap = A + (size_t)(r0 + mr) * K + half * 16;

  if (wid == 0 && lane == 0) tdm_issue_b_tile(btile, K, &sB[0][0]);

  v8f acc[4] = {};
  const int NC = K / KCz;
  for (int c = 0; c < NC; ++c) {
    if (wid == 0) __builtin_amdgcn_s_wait_tensorcnt(0);  // chunk c landed
    __syncthreads();
    if (wid == 0 && lane == 0 && c + 1 < NC)             // overlap next DMA
      tdm_issue_b_tile(btile + (size_t)(c + 1) * KCz, K, &sB[(c + 1) & 1][0]);
    const __bf16* bbase = &sB[c & 1][0] + (size_t)mr * KCz + half * 16;
    const __bf16* apc = ap + (size_t)c * KCz;
#pragma unroll
    for (int k0 = 0; k0 < KCz; k0 += 32) {
      v16bf fa = ld_frag16(apc + k0, apc + k0 + 8);
      const __bf16* b0 = bbase + k0;
      const __bf16* b1 = bbase + 16 * KCz + k0;
      const __bf16* b2 = bbase + 32 * KCz + k0;
      const __bf16* b3 = bbase + 48 * KCz + k0;
      v16bf fb0 = ld_frag16(b0, b0 + 8);
      v16bf fb1 = ld_frag16(b1, b1 + 8);
      v16bf fb2 = ld_frag16(b2, b2 + 8);
      v16bf fb3 = ld_frag16(b3, b3 + 8);
      acc[0] = __builtin_amdgcn_wmma_f32_16x16x32_bf16(false, fa, false, fb0, (short)0, acc[0], false, false);
      acc[1] = __builtin_amdgcn_wmma_f32_16x16x32_bf16(false, fa, false, fb1, (short)0, acc[1], false, false);
      acc[2] = __builtin_amdgcn_wmma_f32_16x16x32_bf16(false, fa, false, fb2, (short)0, acc[2], false, false);
      acc[3] = __builtin_amdgcn_wmma_f32_16x16x32_bf16(false, fa, false, fb3, (short)0, acc[3], false, false);
    }
    __syncthreads();  // all waves done with sB[c&1] before it is re-DMA'd
  }
  int n = lane & 15;
#pragma unroll
  for (int t = 0; t < 4; ++t)
#pragma unroll
    for (int j = 0; j < 8; ++j) {
      int mrow = j + half * 8;
      C[(size_t)(r0 + mrow) * N + c0 + t * 16 + n] = acc[t][j];
    }
}

// ---------------- RoPE + relayout ----------------
__global__ void k_rope(const float* __restrict__ qf, const float* __restrict__ kf,
                       const float* __restrict__ vf,
                       __bf16* __restrict__ qb, __bf16* __restrict__ kb,
                       __bf16* __restrict__ vt) {
  int tid = blockIdx.x * 256 + threadIdx.x;       // over B*NH*S*64 = 2^21
  int d = tid & 63;
  int s = (tid >> 6) & (Sz - 1);
  int h = (tid >> 16) & (NHz - 1);
  int b = tid >> 20;
  float invf = __expf(-((float)(2 * d) / (float)HDz) * 9.210340371976184f); // ln(1e4)
  float ang = (float)s * invf;
  float sn, c;
  __sincosf(ang, &sn, &c);
  size_t rowoff = ((size_t)(b * Sz + s)) * HIDz + h * HDz;
  float q1 = qf[rowoff + d], q2 = qf[rowoff + d + 64];
  float k1 = kf[rowoff + d], k2 = kf[rowoff + d + 64];
  size_t hoff = ((size_t)(b * NHz + h) * Sz + s) * HDz;
  qb[hoff + d]      = (__bf16)(q1 * c - q2 * sn);
  qb[hoff + d + 64] = (__bf16)(q2 * c + q1 * sn);
  kb[hoff + d]      = (__bf16)(k1 * c - k2 * sn);
  kb[hoff + d + 64] = (__bf16)(k2 * c + k1 * sn);
  size_t voff = ((size_t)(b * NHz + h)) * (size_t)HDz * Sz;
  vt[voff + (size_t)d * Sz + s]        = (__bf16)vf[rowoff + d];
  vt[voff + (size_t)(d + 64) * Sz + s] = (__bf16)vf[rowoff + d + 64];
}

// ---------------- attention scores ----------------
__global__ void k_scores(const __bf16* __restrict__ q, const __bf16* __restrict__ k,
                         float* __restrict__ attn) {
  int bh = blockIdx.z;
  int wid = threadIdx.x >> 5;
  int r0 = blockIdx.y * 128 + wid * 16;
  int c0 = blockIdx.x * 64;
  float* out = attn + (size_t)bh * Sz * Sz;
  v8f acc[4] = {};
  if (c0 <= r0 + 15)
    wmma_strip(q + (size_t)bh * Sz * HDz + (size_t)r0 * HDz,
               k + (size_t)bh * Sz * HDz + (size_t)c0 * HDz, HDz, acc);
  int lane = threadIdx.x & 31, n = lane & 15, half = lane >> 4;
#pragma unroll
  for (int t = 0; t < 4; ++t)
#pragma unroll
    for (int j = 0; j < 8; ++j) {
      int row = r0 + j + half * 8;
      int col = c0 + t * 16 + n;
      out[(size_t)row * Sz + col] = (col <= row) ? acc[t][j] * SCALEc : MASKVAL;
    }
}

// ---------------- row softmax ----------------
__global__ void k_softmax(const float* __restrict__ attn, float* __restrict__ scores) {
  __shared__ float sh[8];
  size_t row = blockIdx.x;
  const float* in = attn + row * Sz;
  float* out = scores + row * Sz;
  int t = threadIdx.x;
  float v[4];
  float mx = -__builtin_inff();
#pragma unroll
  for (int j = 0; j < 4; ++j) { v[j] = in[t + j * 256]; mx = fmaxf(mx, v[j]); }
  mx = block_max_256(mx, sh);
  float e[4], sum = 0.f;
#pragma unroll
  for (int j = 0; j < 4; ++j) { e[j] = __expf(v[j] - mx); sum += e[j]; }
  sum = block_sum_256(sum, sh);
  float inv = 1.0f / sum;
#pragma unroll
  for (int j = 0; j < 4; ++j) out[t + j * 256] = e[j] * inv;
}

// ---------------- sequential heavy-hitter scan ----------------
__global__ void __launch_bounds__(1024) k_scan(const float* __restrict__ scores,
                                               unsigned char* __restrict__ keep) {
  __shared__ float shS[32];
  __shared__ float shV[32];
  __shared__ int   shI[32];
  const int bh = blockIdx.x;
  const int s  = threadIdx.x;
  const float* sc = scores + (size_t)bh * Sz * Sz;
  unsigned char* kp = keep + (size_t)bh * Sz * Sz;
  for (int t = 0; t <= CBz; ++t) kp[(size_t)t * Sz + s] = 1;
  float sel = 0.f;
  for (int t = 0; t < CBz; ++t) sel = sel * FFc + sc[(size_t)t * Sz + s];
  bool m = true;
  for (int t = CBz; t <= Sz - 2; ++t) {
    float x = sc[(size_t)t * Sz + s];
    // pull next step's score row toward the caches while we do ~20 reduction
    // rounds for this step (latency-bound, sequential loop)
    __builtin_prefetch(sc + (size_t)(t + 1) * Sz + s, 0, 1);
    float cur = m ? x : 0.f;
    float sum = block_sum_1024(cur, shS);
    sel = FFc * sel + cur / sum;
    int local = t - RBz;
    float v = (s >= SBz && s <= local) ? sel : __builtin_inff();
    int idx = s;
    block_argmin_1024(v, idx, shV, shI);   // first-min tie break (smaller index)
    if (s == idx) { sel = __builtin_inff(); m = false; }
    kp[(size_t)(t + 1) * Sz + s] = m ? (unsigned char)1 : (unsigned char)0;
  }
}

// ---------------- masked softmax -> bf16 probs ----------------
__global__ void k_mask_softmax(const float* __restrict__ attn,
                               const unsigned char* __restrict__ keep,
                               __bf16* __restrict__ probs) {
  __shared__ float sh[8];
  size_t row = blockIdx.x;
  const float* in = attn + row * Sz;
  const unsigned char* kp = keep + row * Sz;
  __bf16* out = probs + row * Sz;
  int t = threadIdx.x;
  float v[4];
  float mx = -__builtin_inff();
#pragma unroll
  for (int j = 0; j < 4; ++j) {
    int idx = t + j * 256;
    v[j] = kp[idx] ? in[idx] : MASKVAL;
    mx = fmaxf(mx, v[j]);
  }
  mx = block_max_256(mx, sh);
  float e[4], sum = 0.f;
#pragma unroll
  for (int j = 0; j < 4; ++j) { e[j] = __expf(v[j] - mx); sum += e[j]; }
  sum = block_sum_256(sum, sh);
  float inv = 1.0f / sum;
#pragma unroll
  for (int j = 0; j < 4; ++j) out[t + j * 256] = (__bf16)(e[j] * inv);
}

// ---------------- PV ----------------
__global__ void k_pv(const __bf16* __restrict__ probs, const __bf16* __restrict__ vT,
                     __bf16* __restrict__ outb) {
  int bh = blockIdx.z;
  int b = bh >> 4, h = bh & 15;
  int wid = threadIdx.x >> 5;
  int r0 = blockIdx.y * 128 + wid * 16;
  int c0 = blockIdx.x * 64;
  v8f acc[4] = {};
  wmma_strip(probs + (size_t)bh * Sz * Sz + (size_t)r0 * Sz,
             vT + (size_t)bh * HDz * Sz + (size_t)c0 * Sz, Sz, acc);
  int lane = threadIdx.x & 31, n = lane & 15, half = lane >> 4;
#pragma unroll
  for (int t = 0; t < 4; ++t)
#pragma unroll
    for (int j = 0; j < 8; ++j) {
      int srow = r0 + j + half * 8;
      int d = c0 + t * 16 + n;
      outb[((size_t)(b * Sz + srow)) * HIDz + h * HDz + d] = (__bf16)acc[t][j];
    }
}

// ---------------- launcher ----------------
extern "C" void kernel_launch(void* const* d_in, const int* in_sizes, int n_in,
                              void* d_out, int out_size, void* d_ws, size_t ws_size,
                              hipStream_t stream) {
  (void)in_sizes; (void)n_in; (void)out_size; (void)ws_size;
  const float* hs = (const float*)d_in[0];
  const float* wq = (const float*)d_in[1];
  const float* wk = (const float*)d_in[2];
  const float* wv = (const float*)d_in[3];
  const float* wo = (const float*)d_in[4];
  float* out = (float*)d_out;
  char* ws = (char*)d_ws;

  const size_t W_BYTES   = (size_t)HIDz * HIDz * 2;          // 8 MiB
  const size_t ACT_BYTES = (size_t)Bz * Sz * HIDz * 2;       // 8 MiB
  const size_t F32_BYTES = (size_t)Bz * Sz * HIDz * 4;       // 16 MiB
  const size_t ATT_BYTES = (size_t)Bz * NHz * Sz * Sz * 4;   // 128 MiB
  const size_t KEEP_BYTES = (size_t)Bz * NHz * Sz * Sz;      // 32 MiB

  size_t o = 0;
  __bf16* WqT  = (__bf16*)(ws + o); o += W_BYTES;
  __bf16* WkT  = (__bf16*)(ws + o); o += W_BYTES;
  __bf16* WvT  = (__bf16*)(ws + o); o += W_BYTES;
  __bf16* WoT  = (__bf16*)(ws + o); o += W_BYTES;
  __bf16* Xb   = (__bf16*)(ws + o); o += ACT_BYTES;
  __bf16* qb   = (__bf16*)(ws + o); o += ACT_BYTES;
  __bf16* kb   = (__bf16*)(ws + o); o += ACT_BYTES;
  __bf16* vtb  = (__bf16*)(ws + o); o += ACT_BYTES;
  __bf16* aob  = (__bf16*)(ws + o); o += ACT_BYTES;
  float*  qf   = (float*)(ws + o); o += F32_BYTES;
  float*  kf   = (float*)(ws + o); o += F32_BYTES;
  float*  vf   = (float*)(ws + o); o += F32_BYTES;
  unsigned char* keep = (unsigned char*)(ws + o); o += KEEP_BYTES;
  float*  scores = (float*)(ws + o); o += ATT_BYTES;  // later aliased by bf16 probs
  float*  attn   = (float*)(ws + o); o += ATT_BYTES;
  __bf16* probs = (__bf16*)scores;  // scores dead after scan; reuse as bf16 probs

  const int elemsW = HIDz * HIDz;               // 4M
  const int elemsX = Bz * Sz * HIDz;            // 4M
  const int elemsR = Bz * NHz * Sz * (HDz / 2); // 2M
  const int rows   = Bz * NHz * Sz;             // 32768

  k_cvt_transpose<<<elemsW / 256, 256, 0, stream>>>(wq, WqT);
  k_cvt_transpose<<<elemsW / 256, 256, 0, stream>>>(wk, WkT);
  k_cvt_transpose<<<elemsW / 256, 256, 0, stream>>>(wv, WvT);
  k_cvt_transpose<<<elemsW / 256, 256, 0, stream>>>(wo, WoT);
  k_cvt<<<elemsX / 256, 256, 0, stream>>>(hs, Xb);
  dim3 gproj(HIDz / 64, (Bz * Sz) / 128);
  k_gemm<<<gproj, 256, 0, stream>>>(Xb, WqT, qf, HIDz, HIDz);
  k_gemm<<<gproj, 256, 0, stream>>>(Xb, WkT, kf, HIDz, HIDz);
  k_gemm<<<gproj, 256, 0, stream>>>(Xb, WvT, vf, HIDz, HIDz);
  k_rope<<<elemsR / 256, 256, 0, stream>>>(qf, kf, vf, qb, kb, vtb);
  k_scores<<<dim3(Sz / 64, Sz / 128, Bz * NHz), 256, 0, stream>>>(qb, kb, attn);
  k_softmax<<<rows, 256, 0, stream>>>(attn, scores);
  k_scan<<<Bz * NHz, 1024, 0, stream>>>(scores, keep);
  k_mask_softmax<<<rows, 256, 0, stream>>>(attn, keep, probs);
  k_pv<<<dim3(HDz / 64, Sz / 128, Bz * NHz), 256, 0, stream>>>(probs, vtb, aob);
  k_gemm<<<gproj, 256, 0, stream>>>(aob, WoT, out, HIDz, HIDz);
}
